// Attention_26061861552386
// MI455X (gfx1250) — matrix operand order repaired and checked by reading, not compiled, not run on hardware
//
#include <hip/hip_runtime.h>
#include <hip/hip_bf16.h>

// ---------------------------------------------------------------------------
// Multi-head attention forward for MI455X (gfx1250, wave32, WMMA f16->f32).
// B=8, S=2048, H=768, NH=12, HD=64.
// Pipeline: cvt(f32->f16, weights transposed) -> QKV GEMM (WMMA, V written
// pre-transposed) -> flash attention (WMMA + TDM tensor_load_to_lds) ->
// output GEMM (WMMA, f32 out).
// ---------------------------------------------------------------------------

typedef __attribute__((ext_vector_type(16))) _Float16 v16h;
typedef __attribute__((ext_vector_type(8)))  _Float16 v8h;
typedef __attribute__((ext_vector_type(8)))  float    v8f;
typedef __attribute__((ext_vector_type(4)))  unsigned v4u;
typedef __attribute__((ext_vector_type(8)))  unsigned v8u;

union F16x16 { v16h v; v8h h[2]; };

static __device__ inline v8f wmma_f16(v16h a, v16h b, v8f c) {
  // D = A(16x32 f16) * B(32x16 f16) + C(16x16 f32)
  return __builtin_amdgcn_wmma_f32_16x16x32_f16(
      false, a, false, b, (short)0, c, false, false);
}

#define Bsz 8
#define Ssz 2048
#define Hsz 768
#define NHsz 12
#define HDsz 64
#define Msz (Bsz * Ssz)          // 16384 token rows

// ---------------------------------------------------------------------------
// Kernel 0a: fp32 -> fp16 conversion (hidden states)
// ---------------------------------------------------------------------------
__global__ void cvt_f32_f16(const float* __restrict__ in,
                            _Float16* __restrict__ out, int n) {
  int i = blockIdx.x * blockDim.x + threadIdx.x;
  if (i < n) out[i] = (_Float16)in[i];
}

// Kernel 0b: fp32 [K][N] -> fp16 transposed [N][K] (weights, once)
__global__ void cvt_transpose_f16(const float* __restrict__ in,
                                  _Float16* __restrict__ out) {
  int i = blockIdx.x * blockDim.x + threadIdx.x;
  if (i < Hsz * Hsz) {
    int n = i / Hsz, k = i % Hsz;
    out[i] = (_Float16)in[k * Hsz + n];      // coalesced writes
  }
}

// ---------------------------------------------------------------------------
// Kernel 1/3: tiled GEMM  out[M,N] = A[M,K] * W[K,N] + bias, W given as
// Wt[N][K].  Block tile 128x128, 256 threads (8 waves 4x2), wave tile 32x64.
//   mode 0: fp16 out scattered into [B,NH,S,HD]   (Q, K)
//   mode 2: fp16 out scattered into [B,NH,HD,S]   (V, pre-transposed)
//   mode 1: fp32 out row-major [M,N]              (final projection)
// ---------------------------------------------------------------------------
#define LDA 40   // 32 + 8 pad (f16 units); keeps 16B alignment
#define LDB 40

__global__ __launch_bounds__(256) void gemm_wmma(
    const _Float16* __restrict__ A, const _Float16* __restrict__ Wt,
    const float* __restrict__ bias, _Float16* __restrict__ outH,
    float* __restrict__ outF, int mode) {
  __shared__ alignas(16) _Float16 sA[128 * LDA];   // [m][k]
  __shared__ alignas(16) _Float16 sB[128 * LDB];   // [n][k]

  const int tid  = threadIdx.x;
  const int wave = tid >> 5;
  const int lane = tid & 31;
  const int l16  = lane & 15;
  const int hi   = lane >> 4;
  const int wm   = wave >> 1;         // 0..3
  const int wn   = wave & 1;          // 0..1
  const int m0   = blockIdx.x * 128;
  const int n0   = blockIdx.y * 128;

  v8f acc[2][4];
#pragma unroll
  for (int i = 0; i < 2; ++i)
#pragma unroll
    for (int j = 0; j < 4; ++j) acc[i][j] = (v8f){};

  const int srow = tid >> 1;            // staging row 0..127
  const int skk  = (tid & 1) * 16;      // staging k offset 0/16

  for (int kb = 0; kb < Hsz; kb += 32) {
    __syncthreads();
    // --- stage A tile [m][k] and B tile [n][k]; both contiguous b128 ---
    {
      const _Float16* srcA = A + (size_t)(m0 + srow) * Hsz + kb + skk;
      *(v8h*)&sA[srow * LDA + skk]     = *(const v8h*)srcA;
      *(v8h*)&sA[srow * LDA + skk + 8] = *(const v8h*)(srcA + 8);
      const _Float16* srcB = Wt + (size_t)(n0 + srow) * Hsz + kb + skk;
      *(v8h*)&sB[srow * LDB + skk]     = *(const v8h*)srcB;
      *(v8h*)&sB[srow * LDB + skk + 8] = *(const v8h*)(srcB + 8);
    }
    if (kb + 32 < Hsz) {   // prefetch next k-tile (global_prefetch_b8)
      __builtin_prefetch(A  + (size_t)(m0 + srow) * Hsz + kb + 32 + skk, 0, 0);
      __builtin_prefetch(Wt + (size_t)(n0 + srow) * Hsz + kb + 32 + skk, 0, 0);
    }
    __syncthreads();

    // --- fragments ---
    F16x16 af[2];
#pragma unroll
    for (int mt = 0; mt < 2; ++mt) {
      int row = wm * 32 + mt * 16 + l16;
      int kA  = hi * 8;                 // lanes 0-15: K 0..7 / 16..23
      af[mt].h[0] = *(const v8h*)&sA[row * LDA + kA];
      af[mt].h[1] = *(const v8h*)&sA[row * LDA + kA + 16];
    }
    F16x16 bf[4];
#pragma unroll
    for (int nt = 0; nt < 4; ++nt) {
      int col = wn * 64 + nt * 16 + l16;
      int kB  = hi * 8;                 // lanes 0-15: K 0..7 / 16..23
      bf[nt].h[0] = *(const v8h*)&sB[col * LDB + kB];
      bf[nt].h[1] = *(const v8h*)&sB[col * LDB + kB + 16];
    }
#pragma unroll
    for (int mt = 0; mt < 2; ++mt)
#pragma unroll
      for (int nt = 0; nt < 4; ++nt)
        acc[mt][nt] = wmma_f16(af[mt].v, bf[nt].v, acc[mt][nt]);
  }

  // --- epilogue: bias + store ---
#pragma unroll
  for (int mt = 0; mt < 2; ++mt) {
#pragma unroll
    for (int nt = 0; nt < 4; ++nt) {
      int gmb = m0 + wm * 32 + mt * 16 + hi * 8;
      int gn  = n0 + wn * 64 + nt * 16 + l16;
      float bv = bias[gn];
#pragma unroll
      for (int r = 0; r < 8; ++r) {
        float val = acc[mt][nt][r] + bv;
        int gm = gmb + r;
        if (mode == 1) {
          outF[(size_t)gm * Hsz + gn] = val;
        } else {
          int b = gm >> 11, s = gm & 2047;
          int h = gn >> 6,  d = gn & 63;
          if (mode == 0)   // [B,NH,S,HD]
            outH[(((size_t)b * NHsz + h) * Ssz + s) * HDsz + d] = (_Float16)val;
          else             // mode 2: [B,NH,HD,S]
            outH[(((size_t)b * NHsz + h) * HDsz + d) * Ssz + s] = (_Float16)val;
        }
      }
    }
  }
}

// ---------------------------------------------------------------------------
// TDM descriptor helpers (CDNA5 ISA).
// 2D tensor, 2-byte elements, LDS row padding 32 DW data + 4 DW pad -> 72
// halfword stride.  Issued per-wave; tracked by TENSORcnt.
// ---------------------------------------------------------------------------
static __device__ inline void tdm_load_2d(unsigned lds_addr, const void* gptr,
                                          unsigned tensor_d0, unsigned tensor_d1,
                                          unsigned stride_d0) {
  unsigned long long ga = (unsigned long long)(size_t)gptr;
  v4u g0;
  g0[0] = 1u;                                   // count=1, user descriptor
  g0[1] = lds_addr;                             // LDS byte address
  g0[2] = (unsigned)ga;                         // global_addr[31:0]
  g0[3] = (unsigned)((ga >> 32) & 0x01FFFFFFu)  // global_addr[56:32]
          | (2u << 30);                         // type = 2 (image)
  v8u g1;
  g1[0] = (1u << 16)      // data_size = 1 -> 2 bytes
        | (1u << 20)      // pad_enable
        | (4u << 22)      // pad_interval code 4 -> every 32 DWORDs
        | (3u << 25);     // pad_amount  code 3 -> 4 DWORDs (8 halfwords)
  g1[1] = (tensor_d0 & 0xFFFFu) << 16;                    // tensor_dim0 lo
  g1[2] = ((tensor_d0 >> 16) & 0xFFFFu)
        | ((tensor_d1 & 0xFFFFu) << 16);                  // dim0 hi | dim1 lo
  g1[3] = ((tensor_d1 >> 16) & 0xFFFFu)
        | (64u << 16);                                    // dim1 hi | tile_dim0=64
  g1[4] = 64u;                                            // tile_dim1=64, tile_dim2=0
  g1[5] = stride_d0;                                      // tensor_dim0_stride lo
  g1[6] = 0u;
  g1[7] = 0u;
  asm volatile("tensor_load_to_lds %0, %1" :: "s"(g0), "s"(g1) : "memory");
}

// ---------------------------------------------------------------------------
// Kernel 2: flash attention.  grid = (S/64, NH, B), 128 threads (4 waves).
// K tile [key][d] and V^T tile [d][key] staged by the Tensor Data Mover.
// ---------------------------------------------------------------------------
#define SKLD 72   // 64 + 8 pad (matches TDM pad config)
#define SVLD 72
#define SPLD 72

__global__ __launch_bounds__(128) void flash_attn(
    const _Float16* __restrict__ Q, const _Float16* __restrict__ K,
    const _Float16* __restrict__ Vt, _Float16* __restrict__ ctx) {
  __shared__ alignas(16) _Float16 sK[64 * SKLD];        // [key][d]
  __shared__ alignas(16) _Float16 sV[64 * SVLD];        // [d][key]
  __shared__ alignas(16) _Float16 sP[4][16 * SPLD];     // per-wave [q][key]

  const int tid  = threadIdx.x;
  const int wave = tid >> 5;
  const int lane = tid & 31;
  const int l16  = lane & 15;
  const int hi   = lane >> 4;
  const int q0   = blockIdx.x * 64;
  const int h    = blockIdx.y;
  const int b    = blockIdx.z;
  const size_t baseK = ((size_t)(b * NHsz + h)) * Ssz * HDsz;  // [S][HD]
  const size_t baseV = ((size_t)(b * NHsz + h)) * HDsz * Ssz;  // [HD][S]

  const unsigned ldsK = (unsigned)(size_t)(void*)&sK[0];
  const unsigned ldsV = (unsigned)(size_t)(void*)&sV[0];

  // --- Q fragments for this wave's 16 rows (A layout) ---
  F16x16 qf[2];
  {
    int qrow = q0 + wave * 16 + l16;
    const _Float16* qp = Q + baseK + (size_t)qrow * HDsz;
#pragma unroll
    for (int c = 0; c < 2; ++c) {
      int kA = c * 32 + hi * 8;
      qf[c].h[0] = *(const v8h*)(qp + kA);
      qf[c].h[1] = *(const v8h*)(qp + kA + 16);
    }
  }

  float mrow[8], lrow[8];
  v8f O[4];
#pragma unroll
  for (int r = 0; r < 8; ++r) { mrow[r] = -1e30f; lrow[r] = 0.f; }
#pragma unroll
  for (int nt = 0; nt < 4; ++nt) O[nt] = (v8f){};

  for (int kt = 0; kt < Ssz / 64; ++kt) {
    __syncthreads();     // previous tile fully consumed
    if (wave == 0) {
      // K tile: rows = 64 keys, row length 64 d, tensor [S x HD] stride HD
      tdm_load_2d(ldsK, K + baseK + (size_t)kt * 64 * HDsz,
                  HDsz, Ssz, HDsz);
      // V^T tile: rows = 64 d, 64 keys each, tensor [HD x S] stride S
      tdm_load_2d(ldsV, Vt + baseV + (size_t)kt * 64,
                  Ssz, HDsz, Ssz);
      __builtin_amdgcn_s_wait_tensorcnt(0);
    }
    __syncthreads();     // tiles visible to all waves

    // --- scores = Q * K^T (16 q x 64 keys) ---
    v8f Sf[4];
#pragma unroll
    for (int nt = 0; nt < 4; ++nt) {
      v8f s = (v8f){};
#pragma unroll
      for (int c = 0; c < 2; ++c) {
        F16x16 bfr;
        int col = nt * 16 + l16;          // key column
        int kB  = c * 32 + hi * 8;        // d chunk
        bfr.h[0] = *(const v8h*)&sK[col * SKLD + kB];
        bfr.h[1] = *(const v8h*)&sK[col * SKLD + kB + 16];
        s = wmma_f16(qf[c].v, bfr.v, s);
      }
      Sf[nt] = s;
    }
#pragma unroll
    for (int nt = 0; nt < 4; ++nt)
#pragma unroll
      for (int r = 0; r < 8; ++r) Sf[nt][r] *= 0.125f;   // 1/sqrt(64)

    // --- online softmax (row reductions inside 16-lane N groups) ---
    _Float16* pw = &sP[wave][0];
#pragma unroll
    for (int r = 0; r < 8; ++r) {
      float t = fmaxf(fmaxf(Sf[0][r], Sf[1][r]), fmaxf(Sf[2][r], Sf[3][r]));
      t = fmaxf(t, __shfl_xor(t, 1, 32));
      t = fmaxf(t, __shfl_xor(t, 2, 32));
      t = fmaxf(t, __shfl_xor(t, 4, 32));
      t = fmaxf(t, __shfl_xor(t, 8, 32));
      float mn = fmaxf(mrow[r], t);
      float alpha = __expf(mrow[r] - mn);
#pragma unroll
      for (int nt = 0; nt < 4; ++nt) O[nt][r] *= alpha;
      float rs = 0.f;
#pragma unroll
      for (int nt = 0; nt < 4; ++nt) {
        float p = __expf(Sf[nt][r] - mn);
        Sf[nt][r] = p;
        rs += p;
      }
      rs += __shfl_xor(rs, 1, 32);
      rs += __shfl_xor(rs, 2, 32);
      rs += __shfl_xor(rs, 4, 32);
      rs += __shfl_xor(rs, 8, 32);
      lrow[r] = lrow[r] * alpha + rs;
      mrow[r] = mn;
#pragma unroll
      for (int nt = 0; nt < 4; ++nt)
        pw[(hi * 8 + r) * SPLD + nt * 16 + l16] = (_Float16)Sf[nt][r];
    }

    // --- O += P * V  (A = P from LDS, B = V^T from LDS) ---
#pragma unroll
    for (int nt = 0; nt < 4; ++nt) {       // d tile
#pragma unroll
      for (int c = 0; c < 2; ++c) {        // key chunk of 32
        F16x16 pf, vf;
        int kA = c * 32 + hi * 8;
        pf.h[0] = *(const v8h*)&pw[l16 * SPLD + kA];
        pf.h[1] = *(const v8h*)&pw[l16 * SPLD + kA + 16];
        int col = nt * 16 + l16;           // d column
        int kB  = c * 32 + hi * 8;         // key rows
        vf.h[0] = *(const v8h*)&sV[col * SVLD + kB];
        vf.h[1] = *(const v8h*)&sV[col * SVLD + kB + 16];
        O[nt] = wmma_f16(pf.v, vf.v, O[nt]);
      }
    }
  }

  // --- normalize and store ctx fp16 in [B, S, H] layout ---
#pragma unroll
  for (int r = 0; r < 8; ++r) {
    float inv = 1.0f / lrow[r];
    int q = q0 + wave * 16 + hi * 8 + r;
#pragma unroll
    for (int nt = 0; nt < 4; ++nt) {
      int col = h * HDsz + nt * 16 + l16;
      ctx[((size_t)b * Ssz + q) * Hsz + col] = (_Float16)(O[nt][r] * inv);
    }
  }
}

// ---------------------------------------------------------------------------
// Host-side launch
// ---------------------------------------------------------------------------
extern "C" void kernel_launch(void* const* d_in, const int* in_sizes, int n_in,
                              void* d_out, int out_size, void* d_ws,
                              size_t ws_size, hipStream_t stream) {
  (void)in_sizes; (void)n_in; (void)out_size; (void)ws_size;

  const float* hs = (const float*)d_in[0];
  const float* Wq = (const float*)d_in[1];
  const float* bq = (const float*)d_in[2];
  const float* Wk = (const float*)d_in[3];
  const float* bk = (const float*)d_in[4];
  const float* Wv = (const float*)d_in[5];
  const float* bv = (const float*)d_in[6];
  const float* Wo = (const float*)d_in[7];
  const float* bo = (const float*)d_in[8];

  const int NHID = Msz * Hsz;       // 12,582,912
  const int NW   = Hsz * Hsz;       // 589,824

  _Float16* ws   = (_Float16*)d_ws;
  _Float16* Xh   = ws;
  _Float16* Wqh  = Xh  + NHID;      // transposed [N][K]
  _Float16* Wkh  = Wqh + NW;
  _Float16* Wvh  = Wkh + NW;
  _Float16* Woh  = Wvh + NW;
  _Float16* Qh   = Woh + NW;        // [B,NH,S,HD]
  _Float16* Kh   = Qh  + NHID;      // [B,NH,S,HD]
  _Float16* Vth  = Kh  + NHID;      // [B,NH,HD,S]  (pre-transposed)
  _Float16* Ctxh = Vth + NHID;      // [B,S,H]

  // 0) fp32 -> fp16 (weights transposed once)
  cvt_f32_f16<<<(NHID + 255) / 256, 256, 0, stream>>>(hs, Xh, NHID);
  cvt_transpose_f16<<<(NW + 255) / 256, 256, 0, stream>>>(Wq, Wqh);
  cvt_transpose_f16<<<(NW + 255) / 256, 256, 0, stream>>>(Wk, Wkh);
  cvt_transpose_f16<<<(NW + 255) / 256, 256, 0, stream>>>(Wv, Wvh);
  cvt_transpose_f16<<<(NW + 255) / 256, 256, 0, stream>>>(Wo, Woh);

  // 1) Q/K/V projections (fp16 out; V pre-transposed per head)
  dim3 ggrid(Msz / 128, Hsz / 128);  // 128 x 6
  gemm_wmma<<<ggrid, 256, 0, stream>>>(Xh, Wqh, bq, Qh,  nullptr, 0);
  gemm_wmma<<<ggrid, 256, 0, stream>>>(Xh, Wkh, bk, Kh,  nullptr, 0);
  gemm_wmma<<<ggrid, 256, 0, stream>>>(Xh, Wvh, bv, Vth, nullptr, 2);

  // 2) flash attention (TDM-staged tiles)
  dim3 agrid(Ssz / 64, NHsz, Bsz);   // 32 x 12 x 8
  flash_attn<<<agrid, 128, 0, stream>>>(Qh, Kh, Vth, Ctxh);

  // 3) output projection (fp32 out)
  gemm_wmma<<<ggrid, 256, 0, stream>>>(Ctxh, Woh, bo, nullptr,
                                       (float*)d_out, 1);
}
